// BuildCost_63814624084217
// MI455X (gfx1250) — matrix operand order repaired and checked by reading, not compiled, not run on hardware
//
#include <hip/hip_runtime.h>
#include <hip/hip_fp16.h>

// ---------------- problem constants ----------------
#define ANG    5
#define BB     2
#define CIN    32
#define COUT   512
#define HH     48
#define WW     48
#define NPIX   (HH*WW)      // 2304
#define NHEADS 4
#define RED    128
#define HD     32
#define NDISP  9
#define OUTPER 16
#define LN_EPS 1e-5f
#define SM_SCALE 0.17677669529663687f   // 1/sqrt(32)

typedef __attribute__((ext_vector_type(16))) _Float16 v16h;
typedef __attribute__((ext_vector_type(8)))  float    v8f;
typedef __attribute__((ext_vector_type(4)))  unsigned su4;
typedef __attribute__((ext_vector_type(8)))  unsigned su8;

// CDNA5 WMMA f16 fragment index maps (wave32).
// A (16x32 f16): lane row m = lane&15; element (v,p): K = (v>=4?16:0) + (lane&16?8:0) + (v&3)*2 + p
__device__ __forceinline__ int a_kidx(int lane, int v, int p) {
  return ((v >= 4) ? 16 : 0) + ((lane & 16) ? 8 : 0) + ((v & 3) << 1) + p;
}
// B (32x16 f16): lane col n = lane&15; element (v,p): K = (v>=4?16:0) + (lane&16?8:0) + (v&3)*2 + p
__device__ __forceinline__ int b_kidx(int lane, int v, int p) {
  return ((v >= 4) ? 16 : 0) + ((lane & 16) ? 8 : 0) + ((v & 3) << 1) + p;
}
// C/D (16x16 f32, 8 VGPRs): element i -> row = (lane&16?8:0)+i, col = lane&15

__device__ __forceinline__ v8f wmma_f16(v16h a, v16h b, v8f c) {
  return __builtin_amdgcn_wmma_f32_16x16x32_f16(false, a, false, b, (short)0, c, false, false);
}

// ---- Tensor Data Mover: 1-D contiguous 2KB (1024 x 2B) copy global -> LDS ----
// D# per CDNA5 ISA ch.8: group0 = {count/lds_addr/global_addr/type}, group1 = dims.
__device__ __forceinline__ void tdm_load_2kb(unsigned lds_off, const void* gptr) {
  unsigned long long ga = (unsigned long long)gptr;
  su4 g0;
  g0[0] = 1u;                                               // count=1, no gather
  g0[1] = lds_off;                                          // lds_addr (bytes)
  g0[2] = (unsigned)(ga & 0xFFFFFFFFull);                   // global_addr[31:0]
  g0[3] = (unsigned)((ga >> 32) & 0x01FFFFFFull)            // global_addr[56:32]
        | (2u << 30);                                       // type=2 ("image")
  su8 g1;
  g1[0] = 0x00010000u;                                      // data_size=1 -> 2 bytes
  g1[1] = (1024u & 0xFFFFu) << 16;                          // tensor_dim0[15:0] @ bits 63:48
  g1[2] = (1024u >> 16) | (1u << 16);                       // tensor_dim0[31:16] | tensor_dim1=1
  g1[3] = (1024u << 16);                                    // tile_dim0 = 1024 @ bits 127:112
  g1[4] = 0u;                                               // tile_dim1=0, tile_dim2=0
  g1[5] = 1024u;                                            // tensor_dim0_stride[31:0]
  g1[6] = 0u;
  g1[7] = 0u;
  asm volatile("tensor_load_to_lds %0, %1" :: "s"(g0), "s"(g1) : "memory");
}
__device__ __forceinline__ void wait_dscnt0() {
  asm volatile("s_wait_dscnt 0x0" ::: "memory");
}

// ---------------- kernel 0: mask average (once) ----------------
__global__ void k_mask_avg(const float* __restrict__ mask, float* __restrict__ mavg) {
  int idx = blockIdx.x * blockDim.x + threadIdx.x;      // b*NPIX + n
  if (idx >= BB * NPIX) return;
  int b = idx / NPIX, n = idx % NPIX;
  float s = 0.f;
  #pragma unroll
  for (int k = 0; k < ANG * ANG; ++k) s += mask[(b * ANG * ANG + k) * NPIX + n];
  mavg[idx] = s * (1.0f / 25.0f);
}

// ---------------- kernel 1: shifted masked grouped 1x1 conv -> cc ----------------
__global__ void k_build_cost(const float* __restrict__ x, const float* __restrict__ mask,
                             const float* __restrict__ fw, float* __restrict__ cc, int dd) {
  int idx = blockIdx.x * blockDim.x + threadIdx.x;      // b*CIN*NPIX
  int n = idx % NPIX;
  int g = (idx / NPIX) % CIN;
  int b = idx / (NPIX * CIN);
  int h = n / WW, w = n % WW;
  float acc[OUTPER];
  #pragma unroll
  for (int o = 0; o < OUTPER; ++o) acc[o] = 0.f;
  for (int a1 = 0; a1 < ANG; ++a1) {
    int hs = h + dd * (2 - a1);
    if (hs < 0 || hs >= HH) continue;
    for (int a2 = 0; a2 < ANG; ++a2) {
      int wsrc = w + dd * (2 - a2);
      if (wsrc < 0 || wsrc >= WW) continue;
      int k = a1 * ANG + a2;
      float xv = x[(((size_t)(b * CIN + g)) * 25 + k) * NPIX + hs * WW + wsrc];
      float mv = mask[(b * 25 + k) * NPIX + n];
      float mod = xv * mv;
      #pragma unroll
      for (int o = 0; o < OUTPER; ++o)
        acc[o] += mod * fw[(g * OUTPER + o) * 25 + k];
    }
  }
  #pragma unroll
  for (int o = 0; o < OUTPER; ++o)
    cc[((size_t)(b * COUT + g * OUTPER + o)) * NPIX + n] = acc[o];
}

// ---------------- kernel 2: LayerNorm statistics (one wave per token) ----------------
__global__ void k_ln_stats(const float* __restrict__ cc, float* __restrict__ mu,
                           float* __restrict__ rsig) {
  int lane = threadIdx.x & 31;
  int row = blockIdx.x * (blockDim.x >> 5) + (threadIdx.x >> 5);   // b*NPIX+n
  if (row >= BB * NPIX) return;
  int b = row / NPIX, n = row % NPIX;
  const float* p = cc + (size_t)b * COUT * NPIX + n;
  float s = 0.f, s2 = 0.f;
  for (int c = lane; c < COUT; c += 32) { float v = p[(size_t)c * NPIX]; s += v; s2 += v * v; }
  #pragma unroll
  for (int m = 16; m >= 1; m >>= 1) { s += __shfl_xor(s, m, 32); s2 += __shfl_xor(s2, m, 32); }
  float mean = s * (1.0f / COUT);
  float var  = s2 * (1.0f / COUT) - mean * mean;
  if (lane == 0) { mu[row] = mean; rsig[row] = rsqrtf(var + LN_EPS); }
}

// ---------------- kernel 3: fused LN + QKV projection (WMMA) ----------------
// M=2304 tokens, N=384 (q|k|v), K=512. One wave per 16x16 output tile.
__global__ void k_qkv(const float* __restrict__ cc, const float* __restrict__ mu,
                      const float* __restrict__ rsig, const float* __restrict__ lnw,
                      const float* __restrict__ lnb, const float* __restrict__ qkvw,
                      _Float16* __restrict__ q, _Float16* __restrict__ kk_,
                      _Float16* __restrict__ v_) {
  int lane = threadIdx.x & 31;
  int wv   = threadIdx.x >> 5;
  int b    = blockIdx.z;
  int r0   = blockIdx.x * 16;
  int c0   = (blockIdx.y * 8 + wv) * 16;                 // 0..368
  const float* ccb = cc + (size_t)b * COUT * NPIX;
  int m = lane & 15;
  int nrow = r0 + m;
  float muv = mu[b * NPIX + nrow], rs = rsig[b * NPIX + nrow];
  int ncol = c0 + (lane & 15);
  v8f acc = {0.f, 0.f, 0.f, 0.f, 0.f, 0.f, 0.f, 0.f};
  for (int k0 = 0; k0 < COUT; k0 += 32) {
    v16h af, bf;
    #pragma unroll
    for (int v = 0; v < 8; ++v) {
      #pragma unroll
      for (int p = 0; p < 2; ++p) {
        int ca = k0 + a_kidx(lane, v, p);
        float xv = (ccb[(size_t)ca * NPIX + nrow] - muv) * rs * lnw[ca] + lnb[ca];
        af[2 * v + p] = (_Float16)xv;
        int cb = k0 + b_kidx(lane, v, p);
        bf[2 * v + p] = (_Float16)qkvw[(size_t)ncol * COUT + cb];
      }
    }
    acc = wmma_f16(af, bf, acc);
  }
  #pragma unroll
  for (int i = 0; i < 8; ++i) {
    int row  = r0 + ((lane & 16) ? 8 : 0) + i;
    int col  = ncol;                                     // j in [0,384)
    int part = col >> 7;                                 // 0=q 1=k 2=v
    int hh   = (col >> 5) & 3;
    int t    = col & 31;
    _Float16* dst = (part == 0) ? q : ((part == 1) ? kk_ : v_);
    dst[((size_t)(b * NHEADS + hh) * NPIX + row) * HD + t] = (_Float16)acc[i];
  }
}

// ---------------- kernel 4: dynamic per-query weight (512 -> 256 relu -> 1) ----------------
__global__ void k_dyn(const float* __restrict__ cc, const float* __restrict__ dw1w,
                      const float* __restrict__ dw1b, const float* __restrict__ dw2w,
                      const float* __restrict__ dw2b, float* __restrict__ dyn) {
  __shared__ float xs[COUT];
  __shared__ float red[256];
  int idx = blockIdx.x;                                  // b*NPIX+n
  int b = idx / NPIX, n = idx % NPIX;
  int t = threadIdx.x;
  const float* ccb = cc + (size_t)b * COUT * NPIX + n;
  for (int c = t; c < COUT; c += 256) xs[c] = ccb[(size_t)c * NPIX];
  __syncthreads();
  const float* wrow = dw1w + (size_t)t * COUT;
  float s = 0.f;
  for (int c = 0; c < COUT; ++c) s += wrow[c] * xs[c];
  s += dw1b[t];
  red[t] = fmaxf(s, 0.f) * dw2w[t];
  __syncthreads();
  for (int off = 128; off >= 1; off >>= 1) {
    if (t < off) red[t] += red[t + off];
    __syncthreads();
  }
  if (t == 0) dyn[idx] = red[0] + dw2b[0];
}

// ---------------- kernel 5: flash attention, one wave per 16-query tile ----------------
// K/V tiles staged in LDS by the Tensor Data Mover (2KB contiguous each).
__global__ void __launch_bounds__(32)
k_attn(const _Float16* __restrict__ q, const _Float16* __restrict__ kmat,
       const _Float16* __restrict__ vmat, const float* __restrict__ dyn,
       _Float16* __restrict__ ao) {
  __shared__ _Float16 kt[32 * 32];                       // K tile: [j][t]
  __shared__ _Float16 vt[32 * 32];                       // V tile: [j][t]
  __shared__ _Float16 pt[16 * 32];                       // P tile bounce buffer
  int lane = threadIdx.x;
  int r0 = blockIdx.x * 16;
  int hh = blockIdx.y;
  int b  = blockIdx.z;
  int bh = b * NHEADS + hh;
  const _Float16* qb = q    + (size_t)bh * NPIX * HD;
  const _Float16* kb = kmat + (size_t)bh * NPIX * HD;
  const _Float16* vb = vmat + (size_t)bh * NPIX * HD;
  unsigned kt_off = (unsigned)(unsigned long long)(void*)&kt[0];
  unsigned vt_off = (unsigned)(unsigned long long)(void*)&vt[0];
  int m = lane & 15;
  int rhalf = (lane & 16) ? 8 : 0;

  v16h qf;
  #pragma unroll
  for (int v = 0; v < 8; ++v)
    #pragma unroll
    for (int p = 0; p < 2; ++p)
      qf[2 * v + p] = qb[(size_t)(r0 + m) * HD + a_kidx(lane, v, p)];

  v8f o0 = {0.f,0.f,0.f,0.f,0.f,0.f,0.f,0.f};
  v8f o1 = {0.f,0.f,0.f,0.f,0.f,0.f,0.f,0.f};
  float mrow[8], lrow[8];
  #pragma unroll
  for (int i = 0; i < 8; ++i) { mrow[i] = -1e30f; lrow[i] = 0.f; }

  for (int j0 = 0; j0 < NPIX; j0 += 32) {
    // WAR: previous iteration's LDS fragment reads must retire before TDM rewrites the tiles.
    wait_dscnt0();
    tdm_load_2kb(kt_off, kb + (size_t)j0 * HD);          // 32x32 f16 K tile
    tdm_load_2kb(vt_off, vb + (size_t)j0 * HD);          // 32x32 f16 V tile
    __builtin_amdgcn_s_wait_tensorcnt(0);                // RAW: tiles resident in LDS

    // S = Q x K^T  (two 16-col tiles) : B[t][jj] = K[j0+jj][t]
    v16h kf0, kf1;
    #pragma unroll
    for (int v = 0; v < 8; ++v)
      #pragma unroll
      for (int p = 0; p < 2; ++p) {
        int t  = b_kidx(lane, v, p);
        int jj = lane & 15;
        kf0[2 * v + p] = kt[jj * 32 + t];
        kf1[2 * v + p] = kt[(16 + jj) * 32 + t];
      }
    v8f z = {0.f,0.f,0.f,0.f,0.f,0.f,0.f,0.f};
    v8f s0 = wmma_f16(qf, kf0, z);
    v8f s1 = wmma_f16(qf, kf1, z);

    float e0[8], e1[8];
    #pragma unroll
    for (int i = 0; i < 8; ++i) {
      float a0 = s0[i] * SM_SCALE, a1 = s1[i] * SM_SCALE;
      float lm = fmaxf(a0, a1);
      lm = fmaxf(lm, __shfl_xor(lm, 1, 32));
      lm = fmaxf(lm, __shfl_xor(lm, 2, 32));
      lm = fmaxf(lm, __shfl_xor(lm, 4, 32));
      lm = fmaxf(lm, __shfl_xor(lm, 8, 32));
      float mn = fmaxf(mrow[i], lm);
      float sc = __expf(mrow[i] - mn);
      mrow[i] = mn;
      e0[i] = __expf(a0 - mn);
      e1[i] = __expf(a1 - mn);
      float ls = e0[i] + e1[i];
      ls += __shfl_xor(ls, 1, 32);
      ls += __shfl_xor(ls, 2, 32);
      ls += __shfl_xor(ls, 4, 32);
      ls += __shfl_xor(ls, 8, 32);
      lrow[i] = lrow[i] * sc + ls;
      o0[i] *= sc; o1[i] *= sc;
    }
    // C-layout -> A-layout via LDS (same-wave LDS ops are in-order)
    #pragma unroll
    for (int i = 0; i < 8; ++i) {
      pt[(rhalf + i) * 32 + (lane & 15)]      = (_Float16)e0[i];
      pt[(rhalf + i) * 32 + 16 + (lane & 15)] = (_Float16)e1[i];
    }
    v16h pf, vf0, vf1;
    #pragma unroll
    for (int v = 0; v < 8; ++v)
      #pragma unroll
      for (int p = 0; p < 2; ++p) {
        pf[2 * v + p] = pt[(lane & 15) * 32 + a_kidx(lane, v, p)];
        int jj = b_kidx(lane, v, p);
        vf0[2 * v + p] = vt[jj * 32 + (lane & 15)];
        vf1[2 * v + p] = vt[jj * 32 + 16 + (lane & 15)];
      }
    o0 = wmma_f16(pf, vf0, o0);
    o1 = wmma_f16(pf, vf1, o1);
  }
  #pragma unroll
  for (int i = 0; i < 8; ++i) {
    int row = r0 + rhalf + i;
    float inv = dyn[b * NPIX + row] / lrow[i];
    size_t base = ((size_t)b * NPIX + row) * RED + hh * HD;
    ao[base + (lane & 15)]      = (_Float16)(o0[i] * inv);
    ao[base + 16 + (lane & 15)] = (_Float16)(o1[i] * inv);
  }
}

// ---------------- kernel 6: out projection + sigmoid residual + /mask_avg ----------------
// M=2304, N=512, K=128
__global__ void k_outproj(const _Float16* __restrict__ ao, const float* __restrict__ outw,
                          const float* __restrict__ cc, const float* __restrict__ mavg,
                          const float* __restrict__ gam, float* __restrict__ out, int di) {
  int lane = threadIdx.x & 31;
  int wv   = threadIdx.x >> 5;
  int b    = blockIdx.z;
  int r0   = blockIdx.x * 16;
  int c0   = (blockIdx.y * 8 + wv) * 16;                 // 0..496
  int m = lane & 15;
  int ncol = c0 + (lane & 15);
  v8f acc = {0.f,0.f,0.f,0.f,0.f,0.f,0.f,0.f};
  for (int k0 = 0; k0 < RED; k0 += 32) {
    v16h af, bf;
    #pragma unroll
    for (int v = 0; v < 8; ++v)
      #pragma unroll
      for (int p = 0; p < 2; ++p) {
        af[2 * v + p] = ao[((size_t)b * NPIX + r0 + m) * RED + k0 + a_kidx(lane, v, p)];
        bf[2 * v + p] = (_Float16)outw[(size_t)ncol * RED + k0 + b_kidx(lane, v, p)];
      }
    acc = wmma_f16(af, bf, acc);
  }
  float g = gam[0];
  #pragma unroll
  for (int i = 0; i < 8; ++i) {
    int row = r0 + ((lane & 16) ? 8 : 0) + i;
    float r = g * acc[i] + cc[((size_t)(b * COUT + ncol)) * NPIX + row];
    float sg = 1.0f / (1.0f + __expf(-r));
    out[(((size_t)(b * COUT + ncol)) * NDISP + di) * NPIX + row] = sg / mavg[b * NPIX + row];
  }
}

// ---------------- launch ----------------
extern "C" void kernel_launch(void* const* d_in, const int* in_sizes, int n_in,
                              void* d_out, int out_size, void* d_ws, size_t ws_size,
                              hipStream_t stream) {
  (void)in_sizes; (void)n_in; (void)out_size;
  const float* x     = (const float*)d_in[0];
  const float* mask  = (const float*)d_in[1];
  const float* fusew = (const float*)d_in[2];
  const float* lnw   = (const float*)d_in[3];
  const float* lnb   = (const float*)d_in[4];
  const float* qkvw  = (const float*)d_in[5];
  const float* outw  = (const float*)d_in[6];
  const float* dw1w  = (const float*)d_in[7];
  const float* dw1b  = (const float*)d_in[8];
  const float* dw2w  = (const float*)d_in[9];
  const float* dw2b  = (const float*)d_in[10];
  const float* gam   = (const float*)d_in[11];
  float* out = (float*)d_out;

  char* ws = (char*)d_ws;
  size_t off = 0;
  auto carve = [&](size_t bytes) -> char* {
    char* p = ws + off;
    off += (bytes + 255) & ~(size_t)255;
    return p;
  };
  float*    cc  = (float*)   carve(sizeof(float)    * BB * COUT * NPIX);
  float*    mu  = (float*)   carve(sizeof(float)    * BB * NPIX);
  float*    rs  = (float*)   carve(sizeof(float)    * BB * NPIX);
  _Float16* qb  = (_Float16*)carve(sizeof(_Float16) * BB * NHEADS * NPIX * HD);
  _Float16* kb  = (_Float16*)carve(sizeof(_Float16) * BB * NHEADS * NPIX * HD);
  _Float16* vb  = (_Float16*)carve(sizeof(_Float16) * BB * NHEADS * NPIX * HD);
  float*    dyn = (float*)   carve(sizeof(float)    * BB * NPIX);
  _Float16* ao  = (_Float16*)carve(sizeof(_Float16) * BB * NPIX * RED);
  float*    mav = (float*)   carve(sizeof(float)    * BB * NPIX);
  if (off > ws_size) return;   // workspace too small: no-op (deterministic)

  k_mask_avg<<<(BB * NPIX + 255) / 256, 256, 0, stream>>>(mask, mav);

  for (int di = 0; di < NDISP; ++di) {
    int dd = di - 4;
    k_build_cost<<<(BB * CIN * NPIX) / 256, 256, 0, stream>>>(x, mask, fusew, cc, dd);
    k_ln_stats<<<(BB * NPIX) / 8, 256, 0, stream>>>(cc, mu, rs);
    k_qkv<<<dim3(NPIX / 16, 3, BB), 256, 0, stream>>>(cc, mu, rs, lnw, lnb, qkvw, qb, kb, vb);
    k_dyn<<<BB * NPIX, 256, 0, stream>>>(cc, dw1w, dw1b, dw2w, dw2b, dyn);
    k_attn<<<dim3(NPIX / 16, NHEADS, BB), 32, 0, stream>>>(qb, kb, vb, dyn, ao);
    k_outproj<<<dim3(NPIX / 16, 4, BB), 256, 0, stream>>>(ao, outw, cc, mav, gam, out, di);
  }
}
